// MultiHeadAttention_66571993088131
// MI455X (gfx1250) — matrix operand order repaired and checked by reading, not compiled, not run on hardware
//
#include <hip/hip_runtime.h>
#include <hip/hip_bf16.h>

// ---------------------------------------------------------------------------
// MHA forward for MI455X (gfx1250, wave32).
//  Kernel 0: one-shot f32 -> bf16 conversion of x and Wq/Wk/Wv (hw cvt).
//  Kernel 1: QKV projections Y = x @ W^T as pure-bf16 WMMA GEMM (f32 accum),
//            writes Q/K/V bf16 in [b,h,n,d] layout.
//  Kernel 2: causal flash-attention per (b,h,64-query block),
//            bf16 WMMA for QK^T and P@V, f32 online softmax.
// ---------------------------------------------------------------------------

typedef __bf16 v16bf __attribute__((ext_vector_type(16)));
typedef float  v8f   __attribute__((ext_vector_type(8)));

#define B_ 4
#define N_ 2048
#define C_ 1024
#define H_ 16
#define D_ 64

union Frag16 {
  v16bf v;
  unsigned short h[16];
  uint4 q[2];
};

// hardware float->bf16 convert (clang lowers fptrunc natively on gfx1250)
__device__ __forceinline__ unsigned short f2bf(float f) {
  union { __bf16 b; unsigned short u; } cv;
  cv.b = (__bf16)f;
  return cv.u;
}

// ---------------------------------------------------------------------------
// Kernel 0: bulk f32 -> bf16 conversion, 8 elements per thread.
// ---------------------------------------------------------------------------
__global__ __launch_bounds__(256) void cvt_f32_bf16_kernel(
    const float* __restrict__ src, unsigned short* __restrict__ dst, int n8) {
  const int i = blockIdx.x * 256 + threadIdx.x;
  if (i >= n8) return;
  const float4* s = reinterpret_cast<const float4*>(src) + (size_t)i * 2;
  const float4 a = s[0];
  const float4 b = s[1];
  union { uint4 q; __bf16 h[8]; } u;
  u.h[0] = (__bf16)a.x; u.h[1] = (__bf16)a.y;
  u.h[2] = (__bf16)a.z; u.h[3] = (__bf16)a.w;
  u.h[4] = (__bf16)b.x; u.h[5] = (__bf16)b.y;
  u.h[6] = (__bf16)b.z; u.h[7] = (__bf16)b.w;
  reinterpret_cast<uint4*>(dst)[i] = u.q;
}

// ---------------------------------------------------------------------------
// Kernel 1: QKV projection GEMM, all-bf16 inputs.
// Block = 4 waves covering a 64(M) x 64(O) tile (waves share the same W
// columns -> L0/L2 reuse). Tiles: 3 mats * 128 m-blocks * 16 o-tiles = 6144.
// Hot loop: 2 b128 A loads + 4 * (2 b128 B loads + 1 wmma).
// ---------------------------------------------------------------------------
__global__ __launch_bounds__(128) void qkv_gemm_kernel(
    const unsigned short* __restrict__ xb,   // [8192][1024] bf16
    const unsigned short* __restrict__ wb,   // [3][1024][1024] bf16
    unsigned short* __restrict__ qkv) {
  const int lane = threadIdx.x & 31;
  const int wave = threadIdx.x >> 5;
  const int blk  = blockIdx.x;                 // 0..6143
  const int mat  = blk / (128 * 16);
  const int rem  = blk % (128 * 16);
  const int m0   = (rem / 16) * 64 + wave * 16;  // row tile (b*n rows)
  const int o0   = (rem % 16) * 64;              // col tile (one head)

  const unsigned short* W = wb + (size_t)mat * ((size_t)C_ * C_);
  unsigned short* out = qkv + (size_t)mat * ((size_t)B_ * H_ * N_ * D_);

  const int ln15  = lane & 15;
  const int hi    = lane >> 4;
  const int kbase = hi * 8;                 // unused: A/B frags load contiguous K at 16*hi

  v8f acc[4] = {};
  const unsigned short* xrow = xb + (size_t)(m0 + ln15) * C_;

  for (int k0 = 0; k0 < C_; k0 += 32) {
    Frag16 a;                                // A: row m, 16 contiguous K at 16*hi
    a.q[0] = *reinterpret_cast<const uint4*>(xrow + k0 + hi * 16);
    a.q[1] = *reinterpret_cast<const uint4*>(xrow + k0 + hi * 16 + 8);
#pragma unroll
    for (int t = 0; t < 4; ++t) {
      Frag16 bfr;                            // B: col o, 16 contiguous K at 16*hi
      const unsigned short* ws =
          W + (size_t)(o0 + t * 16 + ln15) * C_ + k0 + hi * 16;
      bfr.q[0] = *reinterpret_cast<const uint4*>(ws);
      bfr.q[1] = *reinterpret_cast<const uint4*>(ws + 8);
      acc[t] = __builtin_amdgcn_wmma_f32_16x16x32_bf16(
          false, a.v, false, bfr.v, (short)0, acc[t], false, false);
    }
  }

  // store bf16 to [b, h, n, d]
  const int head = o0 >> 6;
#pragma unroll
  for (int t = 0; t < 4; ++t) {
    const int d = t * 16 + ln15;
#pragma unroll
    for (int r = 0; r < 8; ++r) {
      const int m  = m0 + r + 8 * hi;
      const int bb = m >> 11;            // / 2048
      const int nn = m & 2047;
      const size_t idx = ((size_t)(bb * H_ + head) * N_ + nn) * D_ + d;
      out[idx] = f2bf(acc[t][r]);
    }
  }
}

// ---------------------------------------------------------------------------
// Kernel 2: causal flash attention.
// Block = 4 waves (128 thr); wave w owns queries [qb0+16w, qb0+16w+16).
// Loop over 32-key blocks; K staged row-major in LDS, V staged transposed.
// Grid: (32 q-tiles, 16 heads, 4 batch).
// ---------------------------------------------------------------------------
__global__ __launch_bounds__(128) void attn_kernel(
    const unsigned short* __restrict__ Qb,
    const unsigned short* __restrict__ Kb,
    const unsigned short* __restrict__ Vb,
    float* __restrict__ outp) {
  __shared__ unsigned short Ks[32 * 72];       // K rows  [32][64], stride 72
  __shared__ unsigned short Vt[64 * 40];       // V^T     [64][32], stride 40
  __shared__ unsigned short Pw[4 * 16 * 32];   // per-wave P tile [16][32]

  const int lane  = threadIdx.x & 31;
  const int wave  = threadIdx.x >> 5;
  const int qtile = blockIdx.x;
  const int head  = blockIdx.y;
  const int bidx  = blockIdx.z;
  const int qb0   = qtile * 64;
  const int qbase = qb0 + wave * 16;

  const size_t bh = (size_t)(bidx * H_ + head) * N_ * D_;
  const unsigned short* Q = Qb + bh;
  const unsigned short* K = Kb + bh;
  const unsigned short* V = Vb + bh;

  const int ln15  = lane & 15;
  const int hi    = lane >> 4;
  const int kbase = hi * 8;

  // Q A-fragments (reused across the whole key loop)
  Frag16 qa[2];
#pragma unroll
  for (int c = 0; c < 2; ++c) {
    const unsigned short* qs = Q + (size_t)(qbase + ln15) * D_ + c * 32 + hi * 16;
    qa[c].q[0] = *reinterpret_cast<const uint4*>(qs);
    qa[c].q[1] = *reinterpret_cast<const uint4*>(qs + 8);
  }

  v8f O[4] = {};
  float mi[8], li[8];
#pragma unroll
  for (int r = 0; r < 8; ++r) { mi[r] = -1e30f; li[r] = 0.0f; }

  unsigned short* Pme = Pw + wave * (16 * 32);

  const int jend = qb0 + 64;
  for (int j = 0; j < jend; j += 32) {
    __syncthreads();
    {
      // cooperative staging: thread t -> row r = t/4, 16-col chunk c0
      const int t  = threadIdx.x;
      const int r  = t >> 2;
      const int c0 = (t & 3) * 16;
      const unsigned short* ksrc = K + (size_t)(j + r) * D_ + c0;
      *reinterpret_cast<uint4*>(&Ks[r * 72 + c0])     =
          *reinterpret_cast<const uint4*>(ksrc);
      *reinterpret_cast<uint4*>(&Ks[r * 72 + c0 + 8]) =
          *reinterpret_cast<const uint4*>(ksrc + 8);
      // V staged transposed: Vt[d][key]
      const unsigned short* vsrc = V + (size_t)(j + r) * D_ + c0;
#pragma unroll
      for (int i = 0; i < 16; ++i) Vt[(c0 + i) * 40 + r] = vsrc[i];
      // prefetch next key block while this one is consumed
      if (j + 32 < jend) {
        __builtin_prefetch(K + (size_t)(j + 32 + r) * D_ + c0, 0, 3);
        __builtin_prefetch(V + (size_t)(j + 32 + r) * D_ + c0, 0, 3);
      }
    }
    __syncthreads();

    if (j <= qbase + 15) {   // wave-uniform causal skip
      // ---- S = Q K^T for 2 sub-tiles of 16 keys ----
      v8f S[2] = {};
#pragma unroll
      for (int s = 0; s < 2; ++s) {
#pragma unroll
        for (int c = 0; c < 2; ++c) {
          Frag16 kb;  // B-frag: column = key s*16+ln15; 16 contiguous d at 16*hi
          const unsigned short* ks = &Ks[(s * 16 + ln15) * 72 + c * 32 + hi * 16];
          kb.q[0] = *reinterpret_cast<const uint4*>(ks);
          kb.q[1] = *reinterpret_cast<const uint4*>(ks + 8);
          S[s] = __builtin_amdgcn_wmma_f32_16x16x32_bf16(
              false, qa[c].v, false, kb.v, (short)0, S[s], false, false);
        }
      }

      // ---- mask, online softmax (rows live across 16-lane halves) ----
      float p0[8], p1[8], alpha[8];
#pragma unroll
      for (int r = 0; r < 8; ++r) {
        const int qi = qbase + r + 8 * hi;
        float s0 = S[0][r] * 0.125f;            // 1/sqrt(64)
        float s1 = S[1][r] * 0.125f;
        if (j + ln15 > qi)      s0 = -1e30f;
        if (j + 16 + ln15 > qi) s1 = -1e30f;
        float mx = fmaxf(s0, s1);
        mx = fmaxf(mx, __shfl_xor(mx, 1));
        mx = fmaxf(mx, __shfl_xor(mx, 2));
        mx = fmaxf(mx, __shfl_xor(mx, 4));
        mx = fmaxf(mx, __shfl_xor(mx, 8));
        const float mnew = fmaxf(mi[r], mx);
        alpha[r] = __expf(mi[r] - mnew);
        p0[r] = __expf(s0 - mnew);
        p1[r] = __expf(s1 - mnew);
        float rs = p0[r] + p1[r];
        rs += __shfl_xor(rs, 1);
        rs += __shfl_xor(rs, 2);
        rs += __shfl_xor(rs, 4);
        rs += __shfl_xor(rs, 8);
        li[r] = li[r] * alpha[r] + rs;
        mi[r] = mnew;
      }

      // ---- restage P (C layout -> A layout) through wave-private LDS ----
#pragma unroll
      for (int r = 0; r < 8; ++r) {
        const int m = r + 8 * hi;
        Pme[m * 32 + ln15]      = f2bf(p0[r]);
        Pme[m * 32 + 16 + ln15] = f2bf(p1[r]);
      }
      Frag16 pa;
      {
        const unsigned short* ps = &Pme[ln15 * 32 + hi * 16];
        pa.q[0] = *reinterpret_cast<const uint4*>(ps);
        pa.q[1] = *reinterpret_cast<const uint4*>(ps + 8);
      }

      // ---- O = O*alpha + P @ V ----
#pragma unroll
      for (int t = 0; t < 4; ++t) {
#pragma unroll
        for (int r = 0; r < 8; ++r) O[t][r] = O[t][r] * alpha[r];
        Frag16 vb;  // B-frag: column d = t*16+ln15; 16 contiguous keys at 16*hi
        const unsigned short* vs = &Vt[(t * 16 + ln15) * 40 + hi * 16];
        vb.q[0] = *reinterpret_cast<const uint4*>(vs);
        vb.q[1] = *reinterpret_cast<const uint4*>(vs + 8);
        O[t] = __builtin_amdgcn_wmma_f32_16x16x32_bf16(
            false, pa.v, false, vb.v, (short)0, O[t], false, false);
      }
    }
  }

  // ---- normalize and write out [b, n, h*64] f32 ----
#pragma unroll
  for (int t = 0; t < 4; ++t) {
#pragma unroll
    for (int r = 0; r < 8; ++r) {
      const int qi = qbase + r + 8 * hi;
      const float val = O[t][r] / li[r];
      const size_t idx =
          ((size_t)bidx * N_ + qi) * C_ + head * 64 + t * 16 + ln15;
      outp[idx] = val;
    }
  }
}

// ---------------------------------------------------------------------------
extern "C" void kernel_launch(void* const* d_in, const int* in_sizes, int n_in,
                              void* d_out, int out_size, void* d_ws, size_t ws_size,
                              hipStream_t stream) {
  const float* x  = (const float*)d_in[0];
  const float* Wq = (const float*)d_in[1];
  const float* Wk = (const float*)d_in[2];
  const float* Wv = (const float*)d_in[3];
  float* out = (float*)d_out;

  // workspace layout (bf16 elements):
  //   [0, 3*P)            Q,K,V   (P = 8,388,608 each -> 48 MB)
  //   [3P, 3P + X)        x_bf16  (X = 8,388,608    -> 16 MB)
  //   [3P+X, 3P+X+3*Wn)   W_bf16  (Wn = 1,048,576 each -> 6 MB)
  const size_t P  = (size_t)B_ * H_ * N_ * D_;      // 8,388,608
  const size_t X  = (size_t)B_ * N_ * C_;           // 8,388,608
  const size_t Wn = (size_t)C_ * C_;                // 1,048,576

  unsigned short* qkv = (unsigned short*)d_ws;
  unsigned short* xb  = qkv + 3 * P;
  unsigned short* wbq = xb + X;

  // one-shot f32 -> bf16 conversions (8 elems/thread)
  cvt_f32_bf16_kernel<<<(int)(X / 8 / 256), 256, 0, stream>>>(x, xb, (int)(X / 8));
  cvt_f32_bf16_kernel<<<(int)(Wn / 8 / 256), 256, 0, stream>>>(Wq, wbq, (int)(Wn / 8));
  cvt_f32_bf16_kernel<<<(int)(Wn / 8 / 256), 256, 0, stream>>>(Wk, wbq + Wn, (int)(Wn / 8));
  cvt_f32_bf16_kernel<<<(int)(Wn / 8 / 256), 256, 0, stream>>>(Wv, wbq + 2 * Wn, (int)(Wn / 8));

  // 3 mats * 128 (64-row blocks) * 16 (64-col tiles) = 6144 blocks
  qkv_gemm_kernel<<<6144, 128, 0, stream>>>(xb, wbq, qkv);

  attn_kernel<<<dim3(N_ / 64, H_, B_), 128, 0, stream>>>(
      qkv, qkv + P, qkv + 2 * P, out);
}
